// SelfAttention_15710990368994
// MI455X (gfx1250) — matrix operand order repaired and checked by reading, not compiled, not run on hardware
//
#include <hip/hip_runtime.h>

#define D_MODEL 2048
#define LSEQ    2048
#define NB      2
#define NH      16
#define HD      128
#define NTOK    (NB * LSEQ)   // 4096

typedef __attribute__((ext_vector_type(16))) __bf16 v16bf;
typedef __attribute__((ext_vector_type(8)))  float  v8f;

struct alignas(16) U4 { unsigned int v[4]; };

union Frag {
    v16bf          f;
    U4             q[2];
    unsigned int   u[8];
    unsigned short s[16];
};

static __device__ __forceinline__ unsigned short f2bf(float x) {
    unsigned int u = __float_as_uint(x);
    u += 0x7fffu + ((u >> 16) & 1u);   // round-to-nearest-even
    return (unsigned short)(u >> 16);
}

#define WMMA_BF16(A, B, C) \
    __builtin_amdgcn_wmma_f32_16x16x32_bf16(false, (A).f, false, (B).f, (short)0, (C), false, false)

// ---------------------------------------------------------------------------
// Elementwise fp32 -> bf16
// ---------------------------------------------------------------------------
__global__ void f32_to_bf16_k(const float* __restrict__ in,
                              unsigned short* __restrict__ out, int n) {
    int i = blockIdx.x * 256 + threadIdx.x;
    if (i < n) out[i] = f2bf(in[i]);
}

// ---------------------------------------------------------------------------
// LDS-tiled transpose-convert: in [rows][cols] f32 -> out [cols][rows] bf16.
// 32x32 tile per block (256 thr); both global streams coalesced.
// ---------------------------------------------------------------------------
__global__ __launch_bounds__(256) void transpose_f32_to_bf16_k(
        const float* __restrict__ in, unsigned short* __restrict__ out,
        int rows, int cols) {
    __shared__ unsigned short tile[32][33];  // +1 pad: no bank conflicts
    const int tx = threadIdx.x & 31;
    const int ty = threadIdx.x >> 5;         // 0..7
    const int tc = blockIdx.x * 32;          // col base (input)
    const int tr = blockIdx.y * 32;          // row base (input)
#pragma unroll
    for (int i = 0; i < 32; i += 8)
        tile[ty + i][tx] = f2bf(in[(size_t)(tr + ty + i) * cols + tc + tx]);
    __syncthreads();
#pragma unroll
    for (int i = 0; i < 32; i += 8)
        out[(size_t)(tc + ty + i) * rows + tr + tx] = tile[tx][ty + i];
}

// ---------------------------------------------------------------------------
// bf16 WMMA GEMM:  C[M][N] (f32) = A[M][K] (bf16, row-major) * BT[N][K]^T
// Block = 128 threads = 4 waves; block tile 32M x 256N; wave tile 32M x 64N.
// K-blocks (2 A-frags + 4 B-frags = 12 b128 loads) double-buffered; the
// steady-state loop is branch-free (tail peeled) so the 8 WMMAs of block i
// overlap the 12 in-flight loads of block i+1.
// ---------------------------------------------------------------------------
struct KBlock { Frag a[2]; Frag b[4]; };

static __device__ __forceinline__ void load_block(KBlock& kb,
        const unsigned short* a0, const unsigned short* a1,
        const unsigned short* bbase, const size_t* ntOff, int kk) {
    kb.a[0].q[0] = *(const U4*)(a0 + kk);
    kb.a[0].q[1] = *(const U4*)(a0 + kk + 16);
    kb.a[1].q[0] = *(const U4*)(a1 + kk);
    kb.a[1].q[1] = *(const U4*)(a1 + kk + 16);
#pragma unroll
    for (int nt = 0; nt < 4; ++nt) {
        const unsigned short* p = bbase + ntOff[nt] + kk;
        kb.b[nt].q[0] = *(const U4*)p;
        kb.b[nt].q[1] = *(const U4*)(p + 16);
    }
}

static __device__ __forceinline__ void compute_block(v8f acc[2][4], KBlock& kb) {
#pragma unroll
    for (int nt = 0; nt < 4; ++nt) {
        acc[0][nt] = WMMA_BF16(kb.a[0], kb.b[nt], acc[0][nt]);
        acc[1][nt] = WMMA_BF16(kb.a[1], kb.b[nt], acc[1][nt]);
    }
}

__global__ __launch_bounds__(128) void gemm_bf16_wmma(
        const unsigned short* __restrict__ A,
        const unsigned short* __restrict__ BT,
        float* __restrict__ C, int M, int N, int K) {
    const int lane   = threadIdx.x & 31;
    const int wv     = threadIdx.x >> 5;
    const int lane16 = lane & 15;
    const int half   = lane >> 4;
    const int nbase  = blockIdx.x * 256 + wv * 64;
    const int mbase  = blockIdx.y * 32;

    v8f acc[2][4];
#pragma unroll
    for (int mt = 0; mt < 2; ++mt)
#pragma unroll
        for (int nt = 0; nt < 4; ++nt) acc[mt][nt] = {};

    const unsigned short* a0p = A + (size_t)(mbase + lane16)      * K + half * 8;
    const unsigned short* a1p = A + (size_t)(mbase + 16 + lane16) * K + half * 8;
    const unsigned short* bbase = BT + (size_t)(nbase + lane16) * K + half * 8;
    size_t ntOff[4];
#pragma unroll
    for (int nt = 0; nt < 4; ++nt) ntOff[nt] = (size_t)nt * 16 * K;

    KBlock buf0, buf1;
    load_block(buf0, a0p, a1p, bbase, ntOff, 0);

    // Branch-free steady state (K multiple of 64; K >= 128).
    int kk = 0;
    for (; kk + 64 < K; kk += 64) {
        load_block(buf1, a0p, a1p, bbase, ntOff, kk + 32);
        __builtin_prefetch(a0p + kk + 64, 0, 1);
        __builtin_prefetch(bbase + kk + 64, 0, 1);
        compute_block(acc, buf0);
        load_block(buf0, a0p, a1p, bbase, ntOff, kk + 64);
        compute_block(acc, buf1);
    }
    // Tail: kk == K-64 -> K-steps at kk and kk+32 remain.
    load_block(buf1, a0p, a1p, bbase, ntOff, kk + 32);
    compute_block(acc, buf0);
    compute_block(acc, buf1);

#pragma unroll
    for (int mt = 0; mt < 2; ++mt)
#pragma unroll
        for (int nt = 0; nt < 4; ++nt)
#pragma unroll
            for (int r = 0; r < 8; ++r) {
                int m = mbase + mt * 16 + r + 8 * half;
                int n = nbase + nt * 16 + lane16;
                C[(size_t)m * N + n] = acc[mt][nt][r];
            }
}

// ---------------------------------------------------------------------------
// RMSNorm + RoPE + repack.
// qkv fp32 [B*L][3*D_MODEL]; writes q,k -> [B,H,L,HD] bf16; v -> [B,H,HD,L] bf16
// One block (128 threads) per (b,l,h) head instance.
// ---------------------------------------------------------------------------
__global__ __launch_bounds__(128) void norm_rope_pack_k(
        const float* __restrict__ qkv, const float* __restrict__ pe,
        const float* __restrict__ qs, const float* __restrict__ ks,
        unsigned short* __restrict__ Qo, unsigned short* __restrict__ Ko,
        unsigned short* __restrict__ VTo) {
    __shared__ float red[128];
    const int t  = threadIdx.x;          // d in [0,128)
    const int id = blockIdx.x;           // (b*L + l)*H + h
    const int h  = id & (NH - 1);
    const int bl = id >> 4;              // b*L + l
    const int l  = bl & (LSEQ - 1);
    const int b  = bl >> 11;

    const float* base = qkv + (size_t)bl * (3 * D_MODEL) + h * HD;
    float qv = base[t];
    float kv = base[D_MODEL + t];
    float vv = base[2 * D_MODEL + t];

    red[t] = qv * qv; __syncthreads();
    for (int s = 64; s > 0; s >>= 1) { if (t < s) red[t] += red[t + s]; __syncthreads(); }
    float qss = red[0]; __syncthreads();
    red[t] = kv * kv; __syncthreads();
    for (int s = 64; s > 0; s >>= 1) { if (t < s) red[t] += red[t + s]; __syncthreads(); }
    float kss = red[0];

    float qn = qv * rsqrtf(qss * (1.0f / HD) + 1e-6f) * qs[t];
    float kn = kv * rsqrtf(kss * (1.0f / HD) + 1e-6f) * ks[t];

    // RoPE: pair (2p, 2p+1); out[2p+c] = pe[l][p][c][0]*x[2p] + pe[l][p][c][1]*x[2p+1]
    const int p = t >> 1, c = t & 1;
    const float* pp = pe + (((size_t)l * (HD / 2) + p) * 2 + c) * 2;
    float f0 = pp[0], f1 = pp[1];
    float qpv = __shfl_xor(qn, 1, 32);
    float kpv = __shfl_xor(kn, 1, 32);
    float q0 = c ? qpv : qn, q1 = c ? qn : qpv;
    float k0 = c ? kpv : kn, k1 = c ? kn : kpv;
    float qr = f0 * q0 + f1 * q1;
    float kr = f0 * k0 + f1 * k1;

    const size_t bh = (size_t)(b * NH + h);
    Qo[(bh * LSEQ + l) * HD + t]  = f2bf(qr);
    Ko[(bh * LSEQ + l) * HD + t]  = f2bf(kr);
    VTo[(bh * HD + t) * LSEQ + l] = f2bf(vv);
}

// ---------------------------------------------------------------------------
// Flash attention, one wave per 16-query tile, online softmax over 32-key blocks.
// Q,K: [B*H][L][HD] bf16; VT: [B*H][HD][L] bf16; O: [B*L][D_MODEL] bf16.
// K fragments batched ahead of score WMMAs; V fragments issued before the
// softmax VALU stretch so P@V finds them resident.
// ---------------------------------------------------------------------------
__global__ __launch_bounds__(128) void attn_fwd_wmma(
        const unsigned short* __restrict__ Q,
        const unsigned short* __restrict__ Km,
        const unsigned short* __restrict__ VT,
        unsigned short* __restrict__ O) {
    __shared__ unsigned short ldsP[4][16 * 32];   // per-wave P staging (C->A relayout)
    const int lane   = threadIdx.x & 31;
    const int wv     = threadIdx.x >> 5;
    const int lane16 = lane & 15;
    const int half   = lane >> 4;
    const int gid = blockIdx.x * 4 + wv;          // 0..4095
    const int qt  = gid & (LSEQ / 16 - 1);
    const int bh  = gid >> 7;                     // b*H + h
    const int h   = bh & (NH - 1);
    const int b   = bh >> 4;

    const unsigned short* Qh = Q  + (size_t)bh * LSEQ * HD;
    const unsigned short* Kh = Km + (size_t)bh * LSEQ * HD;
    const unsigned short* Vh = VT + (size_t)bh * HD * LSEQ;

    // Preload Q A-fragments (16 rows x 128 K, reused for every key block)
    Frag aq[4];
    const unsigned short* qrow = Qh + (size_t)(qt * 16 + lane16) * HD;
#pragma unroll
    for (int kc = 0; kc < 4; ++kc) {
        const unsigned short* p = qrow + kc * 32 + half * 8;
        aq[kc].q[0] = *(const U4*)p;
        aq[kc].q[1] = *(const U4*)(p + 16);
    }

    v8f o[8];
#pragma unroll
    for (int dt = 0; dt < 8; ++dt) o[dt] = {};
    float row_max[8], row_sum[8];
#pragma unroll
    for (int r = 0; r < 8; ++r) { row_max[r] = -1e30f; row_sum[r] = 0.f; }

    const float SC = 0.08838834764831845f;  // HD^-0.5

    for (int kb = 0; kb < LSEQ; kb += 32) {
        // ---- batch-load all 8 K B-fragments (32 keys x 128 d) ----
        Frag kf[8];
#pragma unroll
        for (int t = 0; t < 2; ++t)
#pragma unroll
            for (int kc = 0; kc < 4; ++kc) {
                const unsigned short* kp =
                    Kh + (size_t)(kb + t * 16 + lane16) * HD + kc * 32 + half * 8;
                kf[t * 4 + kc].q[0] = *(const U4*)kp;
                kf[t * 4 + kc].q[1] = *(const U4*)(kp + 16);
            }

        // S = Q @ K^T for 32 keys (two 16x16 C tiles)
        v8f s0 = {}, s1 = {};
#pragma unroll
        for (int kc = 0; kc < 4; ++kc) s0 = WMMA_BF16(aq[kc], kf[kc],     s0);
#pragma unroll
        for (int kc = 0; kc < 4; ++kc) s1 = WMMA_BF16(aq[kc], kf[4 + kc], s1);

        // ---- issue all 8 V B-fragments now; softmax below hides them ----
        Frag vf[8];
#pragma unroll
        for (int dt = 0; dt < 8; ++dt) {
            const unsigned short* vp =
                Vh + (size_t)(dt * 16 + lane16) * LSEQ + kb + half * 8;
            vf[dt].q[0] = *(const U4*)vp;
            vf[dt].q[1] = *(const U4*)(vp + 16);
        }

        // Online softmax: row m = r + 8*half lives in lanes n=0..15 of this half.
        unsigned short* Pl = &ldsP[wv][0];
#pragma unroll
        for (int r = 0; r < 8; ++r) {
            float v0 = s0[r] * SC;
            float v1 = s1[r] * SC;
            float mx = fmaxf(v0, v1);
            mx = fmaxf(mx, __shfl_xor(mx, 1, 32));
            mx = fmaxf(mx, __shfl_xor(mx, 2, 32));
            mx = fmaxf(mx, __shfl_xor(mx, 4, 32));
            mx = fmaxf(mx, __shfl_xor(mx, 8, 32));
            float nm   = fmaxf(row_max[r], mx);
            float corr = __expf(row_max[r] - nm);
            row_max[r] = nm;
            float e0 = __expf(v0 - nm);
            float e1 = __expf(v1 - nm);
            float rs = e0 + e1;
            rs += __shfl_xor(rs, 1, 32);
            rs += __shfl_xor(rs, 2, 32);
            rs += __shfl_xor(rs, 4, 32);
            rs += __shfl_xor(rs, 8, 32);
            row_sum[r] = row_sum[r] * corr + rs;
#pragma unroll
            for (int dt = 0; dt < 8; ++dt) o[dt][r] *= corr;
            int m = r + 8 * half;
            Pl[m * 32 + lane16]      = f2bf(e0);
            Pl[m * 32 + 16 + lane16] = f2bf(e1);
        }

        // Re-read P in A-fragment layout (same-wave LDS ops stay in order)
        Frag pf;
        {
            const unsigned short* pr = Pl + lane16 * 32 + half * 8;
            pf.q[0] = *(const U4*)pr;
            pf.q[1] = *(const U4*)(pr + 16);
        }

        // O += P @ V  (V fragments already resident)
#pragma unroll
        for (int dt = 0; dt < 8; ++dt) o[dt] = WMMA_BF16(pf, vf[dt], o[dt]);
    }

#pragma unroll
    for (int dt = 0; dt < 8; ++dt)
#pragma unroll
        for (int r = 0; r < 8; ++r) {
            int m   = r + 8 * half;
            int tok = qt * 16 + m;
            int d   = dt * 16 + lane16;
            float val = o[dt][r] / row_sum[r];
            O[((size_t)b * LSEQ + tok) * D_MODEL + h * HD + d] = f2bf(val);
        }
}

// ---------------------------------------------------------------------------
extern "C" void kernel_launch(void* const* d_in, const int* in_sizes, int n_in,
                              void* d_out, int out_size, void* d_ws, size_t ws_size,
                              hipStream_t stream) {
    const float* x       = (const float*)d_in[0];
    const float* pe      = (const float*)d_in[1];
    const float* w_qkv   = (const float*)d_in[2];
    const float* w_proj  = (const float*)d_in[3];
    const float* q_scale = (const float*)d_in[4];
    const float* k_scale = (const float*)d_in[5];
    float* out = (float*)d_out;

    // Workspace layout (all offsets 16B-aligned): ~208 MB total
    unsigned short* xb     = (unsigned short*)d_ws;                       // x bf16 [4096][2048]
    unsigned short* wqkvT  = xb    + (size_t)NTOK * D_MODEL;              // W_qkv^T bf16 [6144][2048]
    unsigned short* wprojT = wqkvT + (size_t)3 * D_MODEL * D_MODEL;       // W_proj^T bf16 [2048][2048]
    float*          qkv    = (float*)(wprojT + (size_t)D_MODEL * D_MODEL);// qkv f32 [4096][6144]
    unsigned short* qpk    = (unsigned short*)(qkv + (size_t)NTOK * 3 * D_MODEL);
    unsigned short* kpk    = qpk + (size_t)NTOK * D_MODEL;
    unsigned short* vT     = kpk + (size_t)NTOK * D_MODEL;
    unsigned short* aout   = vT  + (size_t)NTOK * D_MODEL;                // attn out bf16 [4096][2048]

    const int nx = NTOK * D_MODEL;                       // 8,388,608
    f32_to_bf16_k<<<(nx + 255) / 256, 256, 0, stream>>>(x, xb, nx);

    // Tiled transposes: coalesced reads AND writes
    transpose_f32_to_bf16_k<<<dim3((3 * D_MODEL) / 32, D_MODEL / 32), 256, 0, stream>>>(
        w_qkv, wqkvT, D_MODEL, 3 * D_MODEL);
    transpose_f32_to_bf16_k<<<dim3(D_MODEL / 32, D_MODEL / 32), 256, 0, stream>>>(
        w_proj, wprojT, D_MODEL, D_MODEL);

    // qkv = x @ W_qkv   (4096 x 6144, K=2048)
    gemm_bf16_wmma<<<dim3((3 * D_MODEL) / 256, NTOK / 32), 128, 0, stream>>>(
        xb, wqkvT, qkv, NTOK, 3 * D_MODEL, D_MODEL);

    // RMSNorm + RoPE + pack
    norm_rope_pack_k<<<NTOK * NH, 128, 0, stream>>>(
        qkv, pe, q_scale, k_scale, qpk, kpk, vT);

    // Flash attention: B*H*(L/16) = 4096 wave tiles, 4 waves/block
    attn_fwd_wmma<<<(NB * NH * (LSEQ / 16)) / 4, 128, 0, stream>>>(
        qpk, kpk, vT, aout);

    // out = attn_out @ W_proj   (4096 x 2048, K=2048)
    gemm_bf16_wmma<<<dim3(D_MODEL / 256, NTOK / 32), 128, 0, stream>>>(
        aout, wprojT, out, NTOK, D_MODEL, D_MODEL);
}
